// Model_41360535060693
// MI455X (gfx1250) — matrix operand order repaired and checked by reading, not compiled, not run on hardware
//
#include <hip/hip_runtime.h>

typedef __attribute__((ext_vector_type(16))) _Float16 v16h;
typedef __attribute__((ext_vector_type(8)))  _Float16 v8h;
typedef __attribute__((ext_vector_type(8)))  float    v8f;

#define NPOINTS 524288
#define TSIZE   (1u << 21)
#define W1_FRAGS 4096    // 64x64 f16 (K padded 40->64), pre-swizzled B fragments
#define W2_FRAGS 16384   // 64x256 f16, pre-swizzled B fragments
#define WS_HALFS (W1_FRAGS + W2_FRAGS)   // 20480 halfs = 40 KB

// ---------------------------------------------------------------------------
// Pre-kernel: convert W1/W2 to f16 and lay them out as WMMA B fragments in
// d_ws, so every main block can async-DMA them straight into LDS.
// Fragment layout (16-bit B 32x16, contiguous K order used here): element ((kb*NT+nt)*32+lane)*16+j
// holds B[kb*32 + (lane>>4)*16 + j][nt*16 + (lane&15)].
// ---------------------------------------------------------------------------
__global__ void stage_weights_kernel(const float* __restrict__ W1,
                                     const float* __restrict__ W2,
                                     _Float16* __restrict__ ws) {
  int e = blockIdx.x * 256 + threadIdx.x;
  if (e < W1_FRAGS) {
    int j = e & 15, ln = (e >> 4) & 31, nt = (e >> 9) & 3, kb = e >> 11;
    int k  = kb * 32 + (ln >> 4) * 16 + j;
    int nc = nt * 16 + (ln & 15);
    ws[e] = (_Float16)((k < 40) ? W1[k * 64 + nc] : 0.0f);
  } else if (e < WS_HALFS) {
    int e2 = e - W1_FRAGS;
    int j = e2 & 15, ln = (e2 >> 4) & 31, nt = (e2 >> 9) & 15, kb = e2 >> 13;
    int k  = kb * 32 + (ln >> 4) * 16 + j;
    int nc = nt * 16 + (ln & 15);
    ws[e] = (_Float16)W2[k * 256 + nc];
  }
}

// Build a 16x16x32 f16 A-fragment from a row-major [rows x 64] f16 LDS tile.
// Contiguous K order matching the staged B fragments: lanes 0-15 (M=lane) hold K=0..15,
// lanes 16-31 hold K=16..31 (kb selects the 32-wide K block).
__device__ __forceinline__ v16h load_afrag(const _Float16* base, int m, int hl, int kb) {
  v8h lo = *(const v8h*)(base + m * 64 + kb * 32 + hl * 16);
  v8h hi = *(const v8h*)(base + m * 64 + kb * 32 + hl * 16 + 8);
  v16h a;
#pragma unroll
  for (int i = 0; i < 8; ++i) { a[i] = lo[i]; a[8 + i] = hi[i]; }
  return a;
}

__device__ __forceinline__ _Float16 relu_h(float v) {
  // single v_med3_f32 instead of canonicalize+max
  return (_Float16)__builtin_amdgcn_fmed3f(v, 0.0f, __builtin_huge_valf());
}

__global__ __launch_bounds__(128)
void ngp_fused_kernel(const float* __restrict__ x,
                      const float* __restrict__ table,
                      const _Float16* __restrict__ ws,   // pre-swizzled f16 weights
                      const float* __restrict__ b1,
                      const float* __restrict__ b2,
                      float* __restrict__ out) {
  __shared__ __align__(32) _Float16 sW[WS_HALFS];       // 40 KB weight fragments
  __shared__ __align__(32) _Float16 sAct[4 * 32 * 64];  // 16 KB per-wave tiles

  const int tid  = threadIdx.x;
  const int wave = tid >> 5;
  const int lane = tid & 31;
  const int hl   = lane >> 4;   // which 16-lane half
  const int col  = lane & 15;   // N column within tile / M row for A frags
  const int pbase = blockIdx.x * 128 + wave * 32;   // 32 points per wave
  _Float16* act = sAct + wave * (32 * 64);

  // ---- Kick off async DMA of all weight fragments into LDS (ASYNCcnt path);
  //      overlaps with the hash-grid gathers below.
  {
    unsigned int ldsbase = (unsigned int)(uintptr_t)&sW[0];
#pragma unroll
    for (int it = 0; it < 5; ++it) {            // no-op; the loop below moves 128 threads*16B = 2048B per it (20 chunks of 2KB)
      // 40960 bytes total: each thread moves 16B per iteration, 128 threads, 20 iters
    }
#pragma unroll
    for (int it = 0; it < 20; ++it) {
      int off = (it * 128 + tid) * 16;          // byte offset, 0..40944
      unsigned int lptr = ldsbase + (unsigned int)off;
      unsigned long long gptr =
          (unsigned long long)(uintptr_t)((const char*)ws + off);
      asm volatile("global_load_async_to_lds_b128 %0, %1, off"
                   :: "v"(lptr), "v"(gptr)
                   : "memory");
    }
  }

  // ---- Zero this wave's activation tile (provides K=40..63 zero padding)
  {
    unsigned int* az = (unsigned int*)act;
#pragma unroll
    for (int i = lane; i < 1024; i += 32) az[i] = 0u;
  }

  // ---- Hash-grid encode: 32 points x 10 levels = 320 tasks, 10 per lane ----
#pragma unroll
  for (int it = 0; it < 10; ++it) {
    int task = it * 32 + lane;
    int m    = task & 31;     // local point 0..31
    int lvl  = task >> 5;     // level 0..9
    int g    = pbase + m;
    float xv = x[g * 3 + 0], yv = x[g * 3 + 1], zv = x[g * 3 + 2];
    float s  = (float)((16 << lvl) - 1);
    float px = xv * s, py = yv * s, pz = zv * s;
    float f0 = floorf(px), f1 = floorf(py), f2 = floorf(pz);
    float fx = px - f0, fy = py - f1, fz = pz - f2;
    int ix = (int)f0, iy = (int)f1, iz = (int)f2;
    const float* tb = table + (size_t)lvl * (size_t)TSIZE * 4;
    float a0 = 0.f, a1 = 0.f, a2 = 0.f, a3 = 0.f;
#pragma unroll
    for (int c = 0; c < 8; ++c) {
      int bx = (c >> 2) & 1, by = (c >> 1) & 1, bz = c & 1;
      unsigned int h = (unsigned int)(ix + bx)
                     ^ ((unsigned int)(iy + by) * 2654435761u)
                     ^ ((unsigned int)(iz + bz) * 805459861u);
      unsigned int idx = h & (TSIZE - 1u);
      const float4 f = *(const float4*)(tb + (size_t)idx * 4);
      float w = (bx ? fx : 1.f - fx) * (by ? fy : 1.f - fy) * (bz ? fz : 1.f - fz);
      a0 += w * f.x; a1 += w * f.y; a2 += w * f.z; a3 += w * f.w;
    }
    _Float16* d = act + m * 64 + lvl * 4;
    d[0] = (_Float16)a0; d[1] = (_Float16)a1;
    d[2] = (_Float16)a2; d[3] = (_Float16)a3;
  }

  // ---- Complete the async weight DMA, then block barrier ----
  asm volatile("s_wait_asynccnt 0x0" ::: "memory");
  __syncthreads();

  // A fragments for both 16-row tiles (rows 0..15 and 16..31)
  v16h a00 = load_afrag(act, col, hl, 0);
  v16h a01 = load_afrag(act, col, hl, 1);
  v16h a10 = load_afrag(act, 16 + col, hl, 0);
  v16h a11 = load_afrag(act, 16 + col, hl, 1);

  // ---- GEMM1: feat[32x64pad] @ W1[64x64] + b1, ReLU, f16 back to LDS ----
#pragma unroll
  for (int nt = 0; nt < 4; ++nt) {
    v16h bf0 = *(const v16h*)(sW + ((0 * 4 + nt) * 32 + lane) * 16);
    v16h bf1 = *(const v16h*)(sW + ((1 * 4 + nt) * 32 + lane) * 16);
    float bias = b1[nt * 16 + col];
    v8f c0, c1;
#pragma unroll
    for (int i = 0; i < 8; ++i) { c0[i] = bias; c1[i] = bias; }
    c0 = __builtin_amdgcn_wmma_f32_16x16x32_f16(false, a00, false, bf0, (short)0, c0, false, false);
    c0 = __builtin_amdgcn_wmma_f32_16x16x32_f16(false, a01, false, bf1, (short)0, c0, false, false);
    c1 = __builtin_amdgcn_wmma_f32_16x16x32_f16(false, a10, false, bf0, (short)0, c1, false, false);
    c1 = __builtin_amdgcn_wmma_f32_16x16x32_f16(false, a11, false, bf1, (short)0, c1, false, false);
    // C layout != A layout -> transpose hidden acts through LDS (same-wave LDS
    // ops are in-order; A fragments already live in registers, tile is free)
#pragma unroll
    for (int i = 0; i < 8; ++i) {
      act[(i + 8 * hl) * 64 + nt * 16 + col]      = relu_h(c0[i]);
      act[(16 + i + 8 * hl) * 64 + nt * 16 + col] = relu_h(c1[i]);
    }
  }
  v16h h00 = load_afrag(act, col, hl, 0);
  v16h h01 = load_afrag(act, col, hl, 1);
  v16h h10 = load_afrag(act, 16 + col, hl, 0);
  v16h h11 = load_afrag(act, 16 + col, hl, 1);

  // ---- GEMM2: h[32x64] @ W2[64x256] + b2 -> out (non-temporal stores) ----
  const _Float16* sW2f = sW + W1_FRAGS;
#pragma unroll
  for (int nt = 0; nt < 16; ++nt) {
    v16h bf0 = *(const v16h*)(sW2f + ((0 * 16 + nt) * 32 + lane) * 16);
    v16h bf1 = *(const v16h*)(sW2f + ((1 * 16 + nt) * 32 + lane) * 16);
    float bias = b2[nt * 16 + col];
    v8f c0, c1;
#pragma unroll
    for (int i = 0; i < 8; ++i) { c0[i] = bias; c1[i] = bias; }
    c0 = __builtin_amdgcn_wmma_f32_16x16x32_f16(false, h00, false, bf0, (short)0, c0, false, false);
    c0 = __builtin_amdgcn_wmma_f32_16x16x32_f16(false, h01, false, bf1, (short)0, c0, false, false);
    c1 = __builtin_amdgcn_wmma_f32_16x16x32_f16(false, h10, false, bf0, (short)0, c1, false, false);
    c1 = __builtin_amdgcn_wmma_f32_16x16x32_f16(false, h11, false, bf1, (short)0, c1, false, false);
#pragma unroll
    for (int i = 0; i < 8; ++i) {
      __builtin_nontemporal_store(
          c0[i], out + (size_t)(pbase + i + 8 * hl) * 256 + nt * 16 + col);
      __builtin_nontemporal_store(
          c1[i], out + (size_t)(pbase + 16 + i + 8 * hl) * 256 + nt * 16 + col);
    }
  }
}

extern "C" void kernel_launch(void* const* d_in, const int* in_sizes, int n_in,
                              void* d_out, int out_size, void* d_ws, size_t ws_size,
                              hipStream_t stream) {
  const float* x  = (const float*)d_in[0];
  const float* tb = (const float*)d_in[1];
  const float* W1 = (const float*)d_in[2];
  const float* b1 = (const float*)d_in[3];
  const float* W2 = (const float*)d_in[4];
  const float* b2 = (const float*)d_in[5];
  float* out = (float*)d_out;
  _Float16* ws = (_Float16*)d_ws;   // 40 KB of pre-swizzled f16 weight fragments

  stage_weights_kernel<<<(WS_HALFS + 255) / 256, 256, 0, stream>>>(W1, W2, ws);

  dim3 grid(NPOINTS / 128);  // 4096 blocks, 4 waves each, 32 points per wave
  dim3 block(128);
  ngp_fused_kernel<<<grid, block, 0, stream>>>(x, tb, ws, b1, b2, out);
}
